// MotifsLikePredictor_28441273434455
// MI455X (gfx1250) — matrix operand order repaired and checked by reading, not compiled, not run on hardware
//
#include <hip/hip_runtime.h>

typedef __attribute__((ext_vector_type(16))) __bf16 bf16x16;
typedef __attribute__((ext_vector_type(8)))  float  f32x8;
typedef __attribute__((ext_vector_type(4)))  float  f32x4;
typedef __attribute__((ext_vector_type(4)))  unsigned int u32x4;

union Frag16 { bf16x16 f; u32x4 q[2]; };
union Pack8  { u32x4 q; __bf16 b[8]; };

static __device__ __forceinline__ f32x8 wmma_bf16(const Frag16& a, const Frag16& b, f32x8 c) {
  return __builtin_amdgcn_wmma_f32_16x16x32_bf16(false, a.f, false, b.f, (short)0, c, false, false);
}

// ---------------- prep: f32 -> bf16 conversions ----------------
__global__ void cvt_bf16_kernel(const float* __restrict__ src, __bf16* __restrict__ dst, int n) {
  int i = blockIdx.x * 256 + threadIdx.x;
  if (i < n) dst[i] = (__bf16)src[i];
}

// W_rc (51x4096) -> bf16 padded to 64x4096 (rows 51..63 = 0)
__global__ void cvt_wrc_kernel(const float* __restrict__ src, __bf16* __restrict__ dst) {
  int i = blockIdx.x * 256 + threadIdx.x;      // 64*4096 threads
  int r = i >> 12, c = i & 4095;
  float v = (r < 51) ? src[r * 4096 + c] : 0.f;
  dst[i] = (__bf16)v;
}

// ---------------- GEMM1: edge_rep[e][c] = bf16(edge_ctx @ W_pe.T + b_pe) ----------------
// D[c][e] orientation: A = W_pe rows (M=c, K=512), B[k][e] = edge_ctx[e][k].
// grid = 8(c-tiles of 128) x 32(e-tiles of 128) = 256 blocks, 256 threads (8 waves).
__global__ __launch_bounds__(256)
void gemm1_edge_rep(const __bf16* __restrict__ Wpe,   // [1024][512]
                    const __bf16* __restrict__ ectx,  // [4096][512]
                    const float*  __restrict__ bpe,   // [1024]
                    __bf16* __restrict__ erep)        // [4096][1024]
{
  const int tid  = threadIdx.x;
  const int lane = tid & 31, w = tid >> 5;
  const int l15  = lane & 15, hi = lane >> 4;
  const int blk  = blockIdx.x;
  const int ct = blk & 7, et = blk >> 3;
  const int cgrp = w & 3, eg = w >> 2;       // wave tile: 32 c x 64 e

  const f32x8 zero8 = {0.f,0.f,0.f,0.f,0.f,0.f,0.f,0.f};
  f32x8 acc[2][4];
  #pragma unroll
  for (int i = 0; i < 2; ++i)
    #pragma unroll
    for (int j = 0; j < 4; ++j) acc[i][j] = zero8;

  const __bf16* arow[2];
  #pragma unroll
  for (int i = 0; i < 2; ++i)
    arow[i] = Wpe + (ct*128 + cgrp*32 + i*16 + l15) * 512;
  const __bf16* brow[4];
  #pragma unroll
  for (int j = 0; j < 4; ++j)
    brow[j] = ectx + (et*128 + eg*64 + j*16 + l15) * 512;

  #pragma unroll 4
  for (int k0 = 0; k0 < 512; k0 += 32) {
    Frag16 a[2], b[4];
    #pragma unroll
    for (int i = 0; i < 2; ++i) {
      a[i].q[0] = *(const u32x4*)(arow[i] + k0 + 8*hi);
      a[i].q[1] = *(const u32x4*)(arow[i] + k0 + 16 + 8*hi);
    }
    #pragma unroll
    for (int j = 0; j < 4; ++j) {
      b[j].q[0] = *(const u32x4*)(brow[j] + k0 + 8*hi);
      b[j].q[1] = *(const u32x4*)(brow[j] + k0 + 16 + 8*hi);
    }
    #pragma unroll
    for (int i = 0; i < 2; ++i)
      #pragma unroll
      for (int j = 0; j < 4; ++j)
        acc[i][j] = wmma_bf16(a[i], b[j], acc[i][j]);
  }

  #pragma unroll
  for (int i = 0; i < 2; ++i) {
    const int cb = ct*128 + cgrp*32 + i*16 + 8*hi;     // 8 consecutive c per lane
    f32x4 b0 = *(const f32x4*)(bpe + cb);
    f32x4 b1 = *(const f32x4*)(bpe + cb + 4);
    #pragma unroll
    for (int j = 0; j < 4; ++j) {
      const int e = et*128 + eg*64 + j*16 + l15;
      Pack8 po;
      #pragma unroll
      for (int v = 0; v < 8; ++v) {
        float bias = (v < 4) ? b0[v] : b1[v - 4];
        po.b[v] = (__bf16)(acc[i][j][v] + bias);
      }
      *(u32x4*)(erep + e*1024 + cb) = po.q;
    }
  }
}

// ---------------- fused GEMM2 + epilogue + GEMM3 ----------------
// One block per 64-edge tile (grid = 512). For each of 32 p-tiles (128 cols of P):
//   GEMM2: D[p][e] = W_pc x gathered(edge_rep)   (K = 1024, bf16 WMMA)
//   epilogue: +b_pc, * union_features (NT stream), bf16-pack -> LDS [64e][128p] (stride 136)
//   GEMM3: D2[r][e] += W_rc_slice x prod_tile    (K = 128, bf16 WMMA), D2 register-resident
// Final: out[e][r] = D2 + b_rc[r] + freq_table[obj[hi]*151+obj[ti]][r]  (NT store)
#define LSTR 136
__global__ __launch_bounds__(256)
void gemm23_fused(const __bf16* __restrict__ Wpc,    // [4096][1024]
                  const __bf16* __restrict__ erep,   // [4096][1024]
                  const __bf16* __restrict__ Wrc,    // [64][4096]
                  const float*  __restrict__ bpc,    // [4096]
                  const float*  __restrict__ unionf, // [32768][4096]
                  const float*  __restrict__ brc,    // [51]
                  const float*  __restrict__ freq,   // [22801][51]
                  const int*    __restrict__ pairs,  // [32768][2]
                  const int*    __restrict__ objp,   // [4096]
                  float* __restrict__ out)           // [32768][51]
{
  __shared__ __attribute__((aligned(16))) __bf16 prod_lds[64 * LSTR];

  const int tid  = threadIdx.x;
  const int lane = tid & 31, w = tid >> 5;
  const int l15  = lane & 15, hi = lane >> 4;
  const int eBase = blockIdx.x * 64;
  const int pgrp = w & 3;       // GEMM2: wave tile 32p x 32e ; GEMM3: rgrp = same bits
  const int eg   = w >> 2;      // 0..1

  // gather sources for the B operand of GEMM2 (fixed for whole block)
  const __bf16* srcH[2];
  const __bf16* srcT[2];
  #pragma unroll
  for (int j = 0; j < 2; ++j) {
    const int e = eBase + eg*32 + j*16 + l15;
    srcH[j] = erep + (size_t)pairs[2*e]     * 1024;
    srcT[j] = erep + (size_t)pairs[2*e + 1] * 1024;
  }
  const __bf16* wrcrow = Wrc + (size_t)(pgrp*16 + l15) * 4096;

  const f32x8 zero8 = {0.f,0.f,0.f,0.f,0.f,0.f,0.f,0.f};
  f32x8 d2[2];
  d2[0] = zero8; d2[1] = zero8;

  #pragma unroll 1
  for (int pt = 0; pt < 32; ++pt) {
    f32x8 acc[2][2];
    acc[0][0] = zero8; acc[0][1] = zero8; acc[1][0] = zero8; acc[1][1] = zero8;

    const __bf16* arow[2];
    #pragma unroll
    for (int i = 0; i < 2; ++i)
      arow[i] = Wpc + (size_t)(pt*128 + pgrp*32 + i*16 + l15) * 1024;

    // K = 0..511 : head half (rows = hi[e]) ; K = 512..1023 : tail half (rows = ti[e])
    #pragma unroll 1
    for (int half = 0; half < 2; ++half) {
      const __bf16* const* bsrc = half ? srcT : srcH;
      const int kend = half ? 1024 : 512;
      #pragma unroll 4
      for (int k0 = half * 512; k0 < kend; k0 += 32) {
        Frag16 a[2], b[2];
        #pragma unroll
        for (int i = 0; i < 2; ++i) {
          a[i].q[0] = *(const u32x4*)(arow[i] + k0 + 8*hi);
          a[i].q[1] = *(const u32x4*)(arow[i] + k0 + 16 + 8*hi);
        }
        #pragma unroll
        for (int j = 0; j < 2; ++j) {
          b[j].q[0] = *(const u32x4*)(bsrc[j] + k0 + 8*hi);
          b[j].q[1] = *(const u32x4*)(bsrc[j] + k0 + 16 + 8*hi);
        }
        #pragma unroll
        for (int i = 0; i < 2; ++i)
          #pragma unroll
          for (int j = 0; j < 2; ++j)
            acc[i][j] = wmma_bf16(a[i], b[j], acc[i][j]);
      }
    }

    // epilogue: +b_pc, * union_features (non-temporal stream), pack bf16 -> LDS tile
    #pragma unroll
    for (int i = 0; i < 2; ++i) {
      const int plocal = pgrp*32 + i*16 + 8*hi;          // 8 consecutive p per lane
      const int pb = pt*128 + plocal;
      f32x4 bp0 = *(const f32x4*)(bpc + pb);
      f32x4 bp1 = *(const f32x4*)(bpc + pb + 4);
      #pragma unroll
      for (int j = 0; j < 2; ++j) {
        const int el = eg*32 + j*16 + l15;
        const float* up = unionf + (size_t)(eBase + el) * 4096 + pb;
        f32x4 u0 = __builtin_nontemporal_load((const f32x4*)up);
        f32x4 u1 = __builtin_nontemporal_load((const f32x4*)(up + 4));
        Pack8 po;
        #pragma unroll
        for (int v = 0; v < 8; ++v) {
          float bias = (v < 4) ? bp0[v] : bp1[v - 4];
          float um   = (v < 4) ? u0[v]  : u1[v - 4];
          po.b[v] = (__bf16)((acc[i][j][v] + bias) * um);
        }
        *(u32x4*)&prod_lds[el * LSTR + plocal] = po.q;
      }
    }
    __syncthreads();

    // GEMM3 partial: D2[r][e] += W_rc[:, pt-block] x prod_tile
    #pragma unroll
    for (int kk = 0; kk < 128; kk += 32) {
      Frag16 a, b[2];
      const __bf16* ap = wrcrow + pt*128 + kk;
      a.q[0] = *(const u32x4*)(ap + 8*hi);
      a.q[1] = *(const u32x4*)(ap + 16 + 8*hi);
      #pragma unroll
      for (int j = 0; j < 2; ++j) {
        const __bf16* bp = &prod_lds[(eg*32 + j*16 + l15) * LSTR + kk + 8*hi];
        b[j].q[0] = *(const u32x4*)bp;
        b[j].q[1] = *(const u32x4*)(bp + 16);
      }
      #pragma unroll
      for (int j = 0; j < 2; ++j)
        d2[j] = wmma_bf16(a, b[j], d2[j]);
    }
    __syncthreads();
  }

  // final: add b_rc + freq_table bias, store (each (e,r) owned by exactly one lane)
  #pragma unroll
  for (int j = 0; j < 2; ++j) {
    const int e = eBase + eg*32 + j*16 + l15;
    const int h = pairs[2*e], t = pairs[2*e + 1];
    const float* fr = freq + (size_t)(objp[h] * 151 + objp[t]) * 51;
    #pragma unroll
    for (int v = 0; v < 8; ++v) {
      const int r = pgrp*16 + 8*hi + v;
      if (r < 51)
        __builtin_nontemporal_store(d2[j][v] + brc[r] + fr[r], out + (size_t)e * 51 + r);
    }
  }
}

// ---------------- launch ----------------
extern "C" void kernel_launch(void* const* d_in, const int* in_sizes, int n_in,
                              void* d_out, int out_size, void* d_ws, size_t ws_size,
                              hipStream_t stream)
{
  (void)in_sizes; (void)n_in; (void)out_size; (void)ws_size;
  const float* edge_ctx = (const float*)d_in[0];
  const float* unionf   = (const float*)d_in[1];
  const float* W_pe     = (const float*)d_in[2];
  const float* b_pe     = (const float*)d_in[3];
  const float* W_pc     = (const float*)d_in[4];
  const float* b_pc     = (const float*)d_in[5];
  const float* W_rc     = (const float*)d_in[6];
  const float* b_rc     = (const float*)d_in[7];
  const float* freq     = (const float*)d_in[8];
  const int*   pairs    = (const int*)d_in[9];
  const int*   objp     = (const int*)d_in[10];
  float* out = (float*)d_out;

  char* ws = (char*)d_ws;
  __bf16* ec_bf   = (__bf16*)(ws + 0);          // 4096*512*2  = 4,194,304
  __bf16* wpe_bf  = (__bf16*)(ws + 4194304);    // 1024*512*2  = 1,048,576
  __bf16* wpc_bf  = (__bf16*)(ws + 5242880);    // 4096*1024*2 = 8,388,608
  __bf16* wrc_bf  = (__bf16*)(ws + 13631488);   // 64*4096*2   =   524,288
  __bf16* erep_bf = (__bf16*)(ws + 14155776);   // 4096*1024*2 = 8,388,608  (total ~22.5 MB)

  cvt_bf16_kernel<<<(4096*512)/256,  256, 0, stream>>>(edge_ctx, ec_bf,  4096*512);
  cvt_bf16_kernel<<<(1024*512)/256,  256, 0, stream>>>(W_pe,     wpe_bf, 1024*512);
  cvt_bf16_kernel<<<(4096*1024)/256, 256, 0, stream>>>(W_pc,     wpc_bf, 4096*1024);
  cvt_wrc_kernel <<<(64*4096)/256,   256, 0, stream>>>(W_rc,     wrc_bf);

  gemm1_edge_rep<<<256, 256, 0, stream>>>(wpe_bf, ec_bf, b_pe, erep_bf);
  gemm23_fused  <<<512, 256, 0, stream>>>(wpc_bf, erep_bf, wrc_bf, b_pc, unionf,
                                          b_rc, freq, pairs, objp, out);
}
